// QuadNet_3435973837332
// MI455X (gfx1250) — compile-verified, matrix operand order checked by reading, not run on hardware
//
#include <hip/hip_runtime.h>
#include <hip/hip_bf16.h>
#include <stdint.h>

// ---------------------------------------------------------------------------
// Types for CDNA5 WMMA (gfx1250, wave32)
// ---------------------------------------------------------------------------
typedef __attribute__((ext_vector_type(16))) __bf16 v16bf;
typedef __attribute__((ext_vector_type(8)))  float  v8f;
typedef __attribute__((ext_vector_type(8)))  unsigned short u16x8;

__device__ __forceinline__ unsigned short f2bf(float f) {
  union { float f; uint32_t u; } v; v.f = f;
  uint32_t u = v.u;
  uint32_t rb = 0x7FFFu + ((u >> 16) & 1u);   // round-to-nearest-even
  return (unsigned short)((u + rb) >> 16);
}
__device__ __forceinline__ float bf2f(unsigned short h) {
  union { uint32_t u; float f; } v; v.u = ((uint32_t)h) << 16;
  return v.f;
}

// ---------------------------------------------------------------------------
// Core GEMM:  D[m, cout] = sum_k Act[m,k] * W[cout,k]  (+bias, +optional res)
// Act, W are bf16 (as ushort), f32 accumulation via v_wmma_f32_16x16x32_bf16.
// One wave computes a 16(m) x 64(cout) strip = 4 WMMA tiles: the A fragment is
// loaded once per k-step and reused across 4 B fragments (4 wmma / 10 b128).
// lda == K(padded), ldw == K(padded). Cout must be a multiple of 64.
// ---------------------------------------------------------------------------
__global__ __launch_bounds__(128)
void gemm_bf16_wmma(const unsigned short* __restrict__ A, int lda,
                    const unsigned short* __restrict__ W, int ldw,
                    const float* __restrict__ bias,
                    const unsigned short* __restrict__ res, int ldr,
                    float* __restrict__ O32, unsigned short* __restrict__ O16,
                    int ldd, int dcol,
                    int K, int mtiles, int cgroups) {
  int wave = blockIdx.x * (blockDim.x >> 5) + (threadIdx.x >> 5);
  if (wave >= mtiles * cgroups) return;
  int mt = wave % mtiles;          // 16-row tile of faces
  int cg = wave / mtiles;          // 64-wide group of output channels

  int lane = threadIdx.x & 31;
  int half = lane >> 4;            // 0: lanes 0-15, 1: lanes 16-31
  int l    = lane & 15;

  // A fragment: row = mt*16 + l, K-groups {half*8..+7, 16+half*8..+7}
  const unsigned short* ap = A + (size_t)(mt * 16 + l) * lda + half * 8;
  // B fragments = W^T tiles: tile t column n = l -> weight row cg*64 + t*16 + l
  const unsigned short* bp = W + (size_t)(cg * 64 + l) * ldw + half * 8;
  const int wstep = 16 * ldw;      // row stride between the 4 cout sub-tiles

  v8f acc0 = {}, acc1 = {}, acc2 = {}, acc3 = {};
  union U { struct { u16x8 lo, hi; } p; v16bf v; };

  for (int k = 0; k < K; k += 32) {
    U ua, ub0, ub1, ub2, ub3;
    ua.p.lo  = *(const u16x8*)(ap + k);
    ua.p.hi  = *(const u16x8*)(ap + k + 16);
    const unsigned short* b = bp + k;
    ub0.p.lo = *(const u16x8*)(b);
    ub0.p.hi = *(const u16x8*)(b + 16);
    ub1.p.lo = *(const u16x8*)(b + wstep);
    ub1.p.hi = *(const u16x8*)(b + wstep + 16);
    ub2.p.lo = *(const u16x8*)(b + 2 * wstep);
    ub2.p.hi = *(const u16x8*)(b + 2 * wstep + 16);
    ub3.p.lo = *(const u16x8*)(b + 3 * wstep);
    ub3.p.hi = *(const u16x8*)(b + 3 * wstep + 16);
    __builtin_prefetch((const void*)(ap + k + 256), 0, 1);
    acc0 = __builtin_amdgcn_wmma_f32_16x16x32_bf16(
        false, ua.v, false, ub0.v, (short)0, acc0, false, false);
    acc1 = __builtin_amdgcn_wmma_f32_16x16x32_bf16(
        false, ua.v, false, ub1.v, (short)0, acc1, false, false);
    acc2 = __builtin_amdgcn_wmma_f32_16x16x32_bf16(
        false, ua.v, false, ub2.v, (short)0, acc2, false, false);
    acc3 = __builtin_amdgcn_wmma_f32_16x16x32_bf16(
        false, ua.v, false, ub3.v, (short)0, acc3, false, false);
  }

  // Epilogue: C/D layout -> VGPR r holds row mt*16 + half*8 + r, col = l
  int rowb = mt * 16 + half * 8;
  v8f accs[4] = {acc0, acc1, acc2, acc3};
#pragma unroll
  for (int t = 0; t < 4; ++t) {
    int col  = cg * 64 + t * 16 + l;
    float bv = bias ? bias[col] : 0.0f;
#pragma unroll
    for (int r = 0; r < 8; ++r) {
      float v = accs[t][r] + bv;
      size_t row = (size_t)(rowb + r);
      if (res) v += bf2f(res[row * ldr + col]);
      if (O32) O32[row * ldd + dcol + col] = v;
      else     O16[row * ldd + dcol + col] = f2bf(v);
    }
  }
}

// ---------------------------------------------------------------------------
// Gather 1: build face input [M, 288] bf16 (272 real cols + 16 zero pad)
// col = j*64 + c (j=vertex 0..3, c=channel), cols 256..271 = f_infos
// ---------------------------------------------------------------------------
__global__ void gather_faces_k(const float* __restrict__ features,
                               const int* __restrict__ faces,
                               const float* __restrict__ finfo,
                               unsigned short* __restrict__ Xf) {
  int m = blockIdx.x;          // one block per face, blockDim = 288
  int c = threadIdx.x;
  float v = 0.0f;
  if (c < 256) {
    int j = c >> 6, ch = c & 63;
    int vtx = faces[m * 4 + j];
    v = features[(size_t)vtx * 64 + ch];
  } else if (c < 272) {
    v = finfo[(size_t)m * 16 + (c - 256)];
  }
  Xf[(size_t)m * 288 + c] = f2bf(v);
}

// ---------------------------------------------------------------------------
// Norm machinery (InstanceNorm / GroupNorm(1)) over f32 staging [M, C]
// ---------------------------------------------------------------------------
__global__ void zero_f32_k(float* p, int n) {
  int i = blockIdx.x * blockDim.x + threadIdx.x;
  if (i < n) p[i] = 0.0f;
}

__global__ void stats_k(const float* __restrict__ X, int C, int Mrows,
                        float* __restrict__ stats) {  // stats[0..C)=sum, [C..2C)=sumsq
  int c = threadIdx.x;                                 // blockDim == C
  float s = 0.0f, s2 = 0.0f;
  for (int m = blockIdx.x; m < Mrows; m += gridDim.x) {
    float x = X[(size_t)m * C + c];
    s += x; s2 += x * x;
  }
  atomicAdd(&stats[c], s);
  atomicAdd(&stats[C + c], s2);
}

__global__ void finalize_in_k(const float* __restrict__ stats,
                              const float* __restrict__ g, const float* __restrict__ be,
                              int C, float invM,
                              float* __restrict__ scale, float* __restrict__ shift) {
  int c = threadIdx.x;                                 // <<<1, C>>>
  float mu  = stats[c] * invM;
  float var = stats[C + c] * invM - mu * mu;
  float rs  = rsqrtf(var + 1e-5f);
  float sc  = rs * g[c];
  scale[c] = sc;
  shift[c] = be[c] - mu * sc;
}

__global__ void finalize_gn_k(const float* __restrict__ stats,
                              const float* __restrict__ g, const float* __restrict__ be,
                              int C, float invMC,
                              float* __restrict__ scale, float* __restrict__ shift) {
  __shared__ float s1[512], s2[512];
  int c = threadIdx.x;                                 // <<<1, C>>>, C power of 2 <=512
  s1[c] = stats[c]; s2[c] = stats[C + c];
  __syncthreads();
  for (int off = C >> 1; off > 0; off >>= 1) {
    if (c < off) { s1[c] += s1[c + off]; s2[c] += s2[c + off]; }
    __syncthreads();
  }
  float mu  = s1[0] * invMC;
  float var = s2[0] * invMC - mu * mu;
  float rs  = rsqrtf(var + 1e-5f);
  float sc  = rs * g[c];
  scale[c] = sc;
  shift[c] = be[c] - mu * sc;
}

// y = act(x*scale[c] + shift[c]); slope=0 -> ReLU, else leaky
__global__ void apply_norm_act_k(const float* __restrict__ X,
                                 const float* __restrict__ scale,
                                 const float* __restrict__ shift,
                                 unsigned short* __restrict__ Y,
                                 int Cmask, size_t total, float slope) {
  size_t stride = (size_t)gridDim.x * blockDim.x;
  for (size_t i = (size_t)blockIdx.x * blockDim.x + threadIdx.x; i < total; i += stride) {
    int c = (int)(i & (size_t)Cmask);
    float v = X[i] * scale[c] + shift[c];
    v = v > 0.0f ? v : v * slope;
    Y[i] = f2bf(v);
  }
}

// ---------------------------------------------------------------------------
// head MLP: x1[n,c] = leaky_relu_0.1( W[16,64] @ features[n,:] + b )
// ---------------------------------------------------------------------------
__global__ void head_mlp_k(const float* __restrict__ features,
                           const float* __restrict__ w, const float* __restrict__ b,
                           float* __restrict__ x1, int Npts) {
  int i = blockIdx.x * blockDim.x + threadIdx.x;
  if (i >= Npts * 16) return;
  int n = i >> 4, c = i & 15;
  const float* f  = features + (size_t)n * 64;
  const float* wr = w + c * 64;
  float s = b[c];
#pragma unroll
  for (int k = 0; k < 64; ++k) s += f[k] * wr[k];
  x1[i] = s > 0.0f ? s : 0.1f * s;
}

// Gather 2: tbuf[m, 128 + c*4 + j] = x1[faces[m,j], c]  (bf16)
__global__ void gather2_k(const float* __restrict__ x1, const int* __restrict__ faces,
                          unsigned short* __restrict__ tbuf, int Mfaces) {
  int i = blockIdx.x * blockDim.x + threadIdx.x;
  if (i >= Mfaces * 64) return;
  int m = i >> 6, r = i & 63;
  int c = r >> 2, j = r & 3;
  int vtx = faces[m * 4 + j];
  tbuf[(size_t)m * 192 + 128 + r] = f2bf(x1[(size_t)vtx * 16 + c]);
}

// tail_out (64->2) + log_softmax -> d_out [M,2] f32
__global__ void tail_out_lsm_k(const unsigned short* __restrict__ T,
                               const float* __restrict__ w, const float* __restrict__ b,
                               float* __restrict__ out, int Mfaces) {
  int m = blockIdx.x * blockDim.x + threadIdx.x;
  if (m >= Mfaces) return;
  float s0 = b[0], s1 = b[1];
  const unsigned short* tr = T + (size_t)m * 64;
#pragma unroll
  for (int k = 0; k < 64; ++k) {
    float x = bf2f(tr[k]);
    s0 += x * w[k];
    s1 += x * w[64 + k];
  }
  float mx  = fmaxf(s0, s1);
  float lse = mx + logf(expf(s0 - mx) + expf(s1 - mx));
  out[(size_t)m * 2 + 0] = s0 - lse;
  out[(size_t)m * 2 + 1] = s1 - lse;
}

// Convert f32 weights [rows, cols] -> bf16 [rows, ld] with zero pad cols..ld
__global__ void cvt_w_k(const float* __restrict__ src, unsigned short* __restrict__ dst,
                        int rows, int cols, int ld) {
  int i = blockIdx.x * blockDim.x + threadIdx.x;
  int total = rows * ld;
  if (i >= total) return;
  int r = i / ld, c = i - r * ld;
  dst[i] = (c < cols) ? f2bf(src[(size_t)r * cols + c]) : (unsigned short)0;
}

// ---------------------------------------------------------------------------
// Host-side orchestration
// ---------------------------------------------------------------------------
extern "C" void kernel_launch(void* const* d_in, const int* in_sizes, int n_in,
                              void* d_out, int out_size, void* d_ws, size_t ws_size,
                              hipStream_t stream) {
  const int M = 150000, N = 150000;
  const float* features = (const float*)d_in[0];
  const int*   faces    = (const int*)d_in[1];
  const float* finfos   = (const float*)d_in[2];

  // ---- workspace carving -------------------------------------------------
  char* ws = (char*)d_ws;
  size_t off = 0;
  auto carve = [&](size_t bytes) -> char* {
    char* p = ws + off;
    off = (off + bytes + 255) & ~(size_t)255;
    return p;
  };
  unsigned short* act0  = (unsigned short*)carve((size_t)M * 512 * 2); // also holds Xf [M,288]
  unsigned short* act1  = (unsigned short*)carve((size_t)M * 512 * 2);
  float*          g32   = (float*)carve((size_t)M * 512 * 4);
  unsigned short* hsave = (unsigned short*)carve((size_t)M * 128 * 2);
  unsigned short* tbuf  = (unsigned short*)carve((size_t)M * 192 * 2);
  float*          x1    = (float*)carve((size_t)N * 16 * 4);
  float*          stats = (float*)carve(512 * 2 * 4);
  float*          scale = (float*)carve(512 * 4);
  float*          shift = (float*)carve(512 * 4);

  // ---- weight conversion to bf16 (K padded to /32) -----------------------
  struct WInfo { int idx, rows, cols, ld; };
  const WInfo wl[14] = {
    {3, 128, 272, 288},   // fe0      (pad 272 -> 288)
    {7, 128, 128, 128},   // fe1
    {11, 256, 128, 128},  // res blk0
    {15, 512, 256, 256},  // res blk1
    {19, 512, 512, 512},  // res blk2
    {23, 256, 512, 512},  // res blk3
    {27, 128, 256, 256},  // res final
    {29, 256, 128, 128},  // fe2_0
    {33, 512, 256, 256},  // fe2_1
    {37, 128, 512, 512},  // fe_out
    {41, 512, 192, 192},  // tail0
    {45, 256, 512, 512},  // tail1
    {49, 128, 256, 256},  // tail2
    {53, 64, 128, 128},   // tail3
  };
  unsigned short* wp[14];
  for (int i = 0; i < 14; ++i) {
    size_t e = (size_t)wl[i].rows * wl[i].ld;
    wp[i] = (unsigned short*)carve(e * 2);
    int tot = (int)e;
    cvt_w_k<<<(tot + 255) / 256, 256, 0, stream>>>(
        (const float*)d_in[wl[i].idx], wp[i], wl[i].rows, wl[i].cols, wl[i].ld);
  }

  // ---- helpers -----------------------------------------------------------
  auto gemm = [&](const unsigned short* A, int lda, int widx, int K, int Cout,
                  const float* bias, const unsigned short* res, int ldr,
                  float* o32, unsigned short* o16, int ldd, int dcol) {
    int mtiles = M / 16, cgroups = Cout / 64;
    long waves = (long)mtiles * cgroups;
    int blocks = (int)((waves + 3) / 4);
    gemm_bf16_wmma<<<blocks, 128, 0, stream>>>(A, lda, wp[widx], K, bias, res, ldr,
                                               o32, o16, ldd, dcol, K, mtiles, cgroups);
  };
  auto norm = [&](int C, const float* g, const float* be, bool gn, float slope,
                  unsigned short* Y) {
    zero_f32_k<<<(2 * C + 255) / 256, 256, 0, stream>>>(stats, 2 * C);
    stats_k<<<1024, C, 0, stream>>>(g32, C, M, stats);
    if (gn)
      finalize_gn_k<<<1, C, 0, stream>>>(stats, g, be, C, 1.0f / ((float)M * (float)C),
                                         scale, shift);
    else
      finalize_in_k<<<1, C, 0, stream>>>(stats, g, be, C, 1.0f / (float)M, scale, shift);
    size_t total = (size_t)M * C;
    apply_norm_act_k<<<4096, 256, 0, stream>>>(g32, scale, shift, Y, C - 1, total, slope);
  };
  auto P = [&](int i) { return (const float*)d_in[i]; };

  // ---- pipeline ----------------------------------------------------------
  // gather face features -> act0 [M,288] bf16
  gather_faces_k<<<M, 288, 0, stream>>>(features, faces, finfos, act0);

  // fe0: 288 -> 128, InstanceNorm + leaky(0.01)
  gemm(act0, 288, 0, 288, 128, P(4), nullptr, 0, g32, nullptr, 128, 0);
  norm(128, P(5), P(6), false, 0.01f, act1);
  // fe1: 128 -> 128, InstanceNorm + leaky(0.01); output saved as h
  gemm(act1, 128, 1, 128, 128, P(8), nullptr, 0, g32, nullptr, 128, 0);
  norm(128, P(9), P(10), false, 0.01f, hsave);

  // PointResNet blocks (GroupNorm + ReLU)
  gemm(hsave, 128, 2, 128, 256, P(12), nullptr, 0, g32, nullptr, 256, 0);
  norm(256, P(13), P(14), true, 0.0f, act0);
  gemm(act0, 256, 3, 256, 512, P(16), nullptr, 0, g32, nullptr, 512, 0);
  norm(512, P(17), P(18), true, 0.0f, act1);
  gemm(act1, 512, 4, 512, 512, P(20), nullptr, 0, g32, nullptr, 512, 0);
  norm(512, P(21), P(22), true, 0.0f, act0);
  gemm(act0, 512, 5, 512, 256, P(24), nullptr, 0, g32, nullptr, 256, 0);
  norm(256, P(25), P(26), true, 0.0f, act1);
  // res final: 256 -> 128 + residual(hsave) -> bf16 act0 [M,128]
  gemm(act1, 256, 6, 256, 128, P(28), hsave, 128, nullptr, act0, 128, 0);

  // fe2: 128 -> 256 -> 512 (InstanceNorm + leaky 0.01)
  gemm(act0, 128, 7, 128, 256, P(30), nullptr, 0, g32, nullptr, 256, 0);
  norm(256, P(31), P(32), false, 0.01f, act1);
  gemm(act1, 256, 8, 256, 512, P(34), nullptr, 0, g32, nullptr, 512, 0);
  norm(512, P(35), P(36), false, 0.01f, act0);

  // head MLP on vertices: [N,16] f32
  head_mlp_k<<<(N * 16 + 255) / 256, 256, 0, stream>>>(features, P(39), P(40), x1, N);

  // fe_out: 512 -> 128, bias, no norm; write bf16 directly into tbuf cols 0..127
  gemm(act0, 512, 9, 512, 128, P(38), nullptr, 0, nullptr, tbuf, 192, 0);
  // second gather into tbuf cols 128..191
  gather2_k<<<(M * 64 + 255) / 256, 256, 0, stream>>>(x1, faces, tbuf, M);

  // tail: 192 -> 512 -> 256 -> 128 -> 64 (InstanceNorm + ReLU)
  gemm(tbuf, 192, 10, 192, 512, P(42), nullptr, 0, g32, nullptr, 512, 0);
  norm(512, P(43), P(44), false, 0.0f, act1);
  gemm(act1, 512, 11, 512, 256, P(46), nullptr, 0, g32, nullptr, 256, 0);
  norm(256, P(47), P(48), false, 0.0f, act0);
  gemm(act0, 256, 12, 256, 128, P(50), nullptr, 0, g32, nullptr, 128, 0);
  norm(128, P(51), P(52), false, 0.0f, act1);
  gemm(act1, 128, 13, 128, 64, P(54), nullptr, 0, g32, nullptr, 64, 0);
  norm(64, P(55), P(56), false, 0.0f, act0);

  // tail_out 64->2 + log_softmax -> d_out
  tail_out_lsm_k<<<(M + 255) / 256, 256, 0, stream>>>(act0, P(57), P(58),
                                                      (float*)d_out, M);
  (void)in_sizes; (void)n_in; (void)out_size; (void)ws_size;
}
